// ImprovedSNN_46763603918991
// MI455X (gfx1250) — compile-verified, matrix operand order checked by reading, not run on hardware
//
#include <hip/hip_runtime.h>
#include <hip/hip_bf16.h>
#include <math.h>

// ---------------- problem constants ----------------
#define BATCH 2048
#define FDIM  512
#define TDIM  100
#define H1    1024
#define H2    512
#define H3    256
#define ODIM  2
#define M_TILE 16            // batch rows per workgroup -> 128 WGs (fills WGPs)
#define NTHREADS 256         // 8 waves (wave32)
#define BN_EPS 1e-5f

typedef __attribute__((ext_vector_type(16))) _Float16 v16h;
typedef __attribute__((ext_vector_type(8)))  _Float16 v8h;
typedef __attribute__((ext_vector_type(8)))  float    v8f;

// ---------------------------------------------------------------------------
// Fragment helpers (CDNA5 WMMA 16x16x32 f16 layouts, ISA 7.12.2)
//   A (16x32, MxK): lane holds row M = lane%16.  Halves j=0..7 -> K = kb+j,
//   j=8..15 -> K = 16+kb+(j-8), where kb = (lane<16 ? 0 : 8).
//   B (32x16, KxN): symmetric, col = lane%16, same K mapping; weights are
//   pre-packed so each lane reads 16 contiguous halves (1 pair of b128s).
//   C/D (16x16 f32): VGPR r, lane l -> M = r + (l<16?0:8), N = l%16.
// ---------------------------------------------------------------------------
__device__ inline v16h load_a_frag(const _Float16* __restrict__ act, int ldk,
                                   int row, int k0, int lane) {
  const _Float16* p = act + row * ldk + k0 + ((lane & 16) ? 8 : 0);
  v8h lo = *(const v8h*)(p);        // K = kb+0 .. kb+7   (ds_load_b128)
  v8h hi = *(const v8h*)(p + 16);   // K = kb+16 .. kb+23 (ds_load_b128)
  v16h a;
#pragma unroll
  for (int j = 0; j < 8; ++j) { a[j] = lo[j]; a[j + 8] = hi[j]; }
  return a;
}

// CDNA5 async global->LDS copy (ASYNCcnt-tracked, no VGPR round trip).
// Generic shared pointers are flat addresses in the LDS aperture; low 32 bits
// are the LDS byte offset (ISA 10.2 aperture mapping).
__device__ inline void async_copy_b128(void* lds_dst, const void* gsrc) {
  asm volatile("global_load_async_to_lds_b128 %0, %1, off"
               :: "v"((unsigned)(unsigned long long)lds_dst),
                  "v"((unsigned long long)gsrc)
               : "memory");
}
__device__ inline void wait_async0() {
  asm volatile("s_wait_asynccnt 0x0" ::: "memory");
}

// LIF state update + spike store for one 16x16 C tile (fragment-ordered state).
__device__ inline void lif_tile(const v8f& c, int nt, int N,
                                _Float16* __restrict__ zOut,
                                float* __restrict__ vSt, float* __restrict__ iSt,
                                int lane, int blockTileBase) {
  const size_t tile = (size_t)blockTileBase + nt;
  float* vp = vSt + tile * 256 + lane * 8;
  float* ip = iSt + tile * 256 + lane * 8;
  v8f v  = *(const v8f*)vp;
  v8f ii = *(const v8f*)ip;
  const int mrow = (lane & 16) ? 8 : 0;
  const int ncol = nt * 16 + (lane & 15);
#pragma unroll
  for (int r = 0; r < 8; ++r) {
    float vd = v[r] * 0.9f + 0.1f * ii[r];       // v + dt*tau_mem*(-v+i)
    float z  = (vd > 1.0f) ? 1.0f : 0.0f;        // spike vs V_TH=1
    v[r]  = (1.0f - z) * vd;                     // reset
    ii[r] = ii[r] * 0.8f + c[r];                 // i*(1-dt*tau_syn) + inp
    zOut[(mrow + r) * N + ncol] = (_Float16)z;
  }
  *(v8f*)vp = v;
  *(v8f*)ip = ii;
}

// Fused GEMM(+bias) + LIF, single m-tile, NB-wide n-blocking:
// one A fragment (2 ds_load_b128) feeds NB WMMAs.
template <int K, int N, int NT_PER_WAVE, int NB>
__device__ void gemm_lif(const _Float16* __restrict__ actIn,
                         _Float16* __restrict__ zOut,
                         const _Float16* __restrict__ wp,
                         const float* __restrict__ bias,
                         float* __restrict__ vSt, float* __restrict__ iSt,
                         int wave, int lane, int blockTileBase) {
  constexpr int KS = K / 32;
  static_assert(NT_PER_WAVE % NB == 0, "n-tile group mismatch");
#pragma unroll
  for (int g = 0; g < NT_PER_WAVE / NB; ++g) {
    const int ntb = wave * NT_PER_WAVE + g * NB;
    v8f c[NB];
#pragma unroll
    for (int u = 0; u < NB; ++u) {
      const float bs = bias[(ntb + u) * 16 + (lane & 15)];
#pragma unroll
      for (int r = 0; r < 8; ++r) c[u][r] = bs;
    }
#pragma unroll 2
    for (int ks = 0; ks < KS; ++ks) {
      v16h a = load_a_frag(actIn, K, (lane & 15), ks * 32, lane);
#pragma unroll
      for (int u = 0; u < NB; ++u) {
        v16h b = *(const v16h*)(wp + ((size_t)(ntb + u) * KS + ks) * 512 + lane * 16);
        c[u] = __builtin_amdgcn_wmma_f32_16x16x32_f16(false, a, false, b,
                                                      (short)0, c[u], false, false);
      }
    }
#pragma unroll
    for (int u = 0; u < NB; ++u)
      lif_tile(c[u], ntb + u, N, zOut, vSt, iSt, lane, blockTileBase);
  }
}

// ---------------------------------------------------------------------------
// Persistent main kernel: one workgroup owns 16 batch rows for all 100 steps.
// ---------------------------------------------------------------------------
__global__ __launch_bounds__(NTHREADS) void snn_main(
    const float* __restrict__ x, const _Float16* __restrict__ xT, int use_xT,
    const _Float16* __restrict__ W1p, const _Float16* __restrict__ W2p,
    const _Float16* __restrict__ W3p, const _Float16* __restrict__ Woutp,
    const float* __restrict__ bias1, const float* __restrict__ bias2,
    const float* __restrict__ bias3, const float* __restrict__ biasOut,
    const float* __restrict__ decay,
    float* v1, float* i1, float* v2, float* i2, float* v3, float* i3,
    float* __restrict__ out) {
  extern __shared__ _Float16 smem[];
  _Float16* sH0 = smem;                  // 16 x 512 (double-buffered)
  _Float16* sH1 = sH0 + M_TILE * FDIM;
  _Float16* sZ1 = sH1 + M_TILE * FDIM;   // 16 x 1024
  _Float16* sZ2 = sZ1 + M_TILE * H1;     // 16 x 512
  _Float16* sZ3 = sZ2 + M_TILE * H2;     // 16 x 256

  const int tid  = threadIdx.x;
  const int lane = tid & 31;
  const int wave = tid >> 5;
  const int b0   = blockIdx.x * M_TILE;

  // Output-layer LIF state + weighted spike accumulator live in wave-0
  // fragment registers (one 16x16 C tile, cols 0..1 valid).
  v8f vo, io, acc;
#pragma unroll
  for (int r = 0; r < 8; ++r) { vo[r] = 0.0f; io[r] = 0.0f; acc[r] = 0.0f; }

  // Prologue: async-stage t=0 input tile.
  if (use_xT) {
    const _Float16* src = xT + (size_t)b0 * FDIM;
    for (int idx = tid * 8; idx < M_TILE * FDIM; idx += NTHREADS * 8)
      async_copy_b128(sH0 + idx, src + idx);
  }

  for (int t = 0; t < TDIM; ++t) {
    const _Float16* sHcur;
    if (use_xT) {
      sHcur = (t & 1) ? sH1 : sH0;
      wait_async0();                       // own async copies done
    } else {
      // Fallback: strided gather from x (B,F,T), t innermost.
      for (int idx = tid; idx < M_TILE * FDIM; idx += NTHREADS) {
        const int m = idx >> 9, f = idx & 511;
        const float* px = x + ((size_t)(b0 + m) * FDIM + f) * TDIM + t;
        sH0[idx] = (_Float16)(*px);
        if (t + 32 < TDIM) __builtin_prefetch(px + 32, 0, 1);
      }
      sHcur = sH0;
    }
    __syncthreads();                       // all waves' tiles visible

    gemm_lif<FDIM, H1, 8, 4>(sHcur, sZ1, W1p, bias1, v1, i1, wave, lane, blockIdx.x * 64);

    // Hide next-step input copy behind GEMM2/3/out (other buffer, ASYNCcnt).
    if (use_xT && t + 1 < TDIM) {
      _Float16* dst = (t & 1) ? sH0 : sH1;
      const _Float16* src = xT + ((size_t)(t + 1) * BATCH + b0) * FDIM;
      for (int idx = tid * 8; idx < M_TILE * FDIM; idx += NTHREADS * 8)
        async_copy_b128(dst + idx, src + idx);
    }
    __syncthreads();
    gemm_lif<H1,   H2, 4, 4>(sZ1, sZ2, W2p, bias2, v2, i2, wave, lane, blockIdx.x * 32);
    __syncthreads();
    gemm_lif<H2,   H3, 2, 2>(sZ2, sZ3, W3p, bias3, v3, i3, wave, lane, blockIdx.x * 16);
    __syncthreads();

    // Output layer as one zero-padded 16-col WMMA tile on wave 0.
    if (wave == 0) {
      v8f co;
      const float bs = biasOut[lane & 15];
#pragma unroll
      for (int r = 0; r < 8; ++r) co[r] = bs;
#pragma unroll
      for (int ks = 0; ks < H3 / 32; ++ks) {
        v16h a = load_a_frag(sZ3, H3, (lane & 15), ks * 32, lane);
        v16h b = *(const v16h*)(Woutp + ks * 512 + lane * 16);
        co = __builtin_amdgcn_wmma_f32_16x16x32_f16(false, a, false, b,
                                                    (short)0, co, false, false);
      }
      const float w = decay[t];            // DECAY^(T-1-t)
#pragma unroll
      for (int r = 0; r < 8; ++r) {
        float vd = vo[r] * 0.9f + 0.1f * io[r];
        float z  = (vd > 1.0f) ? 1.0f : 0.0f;
        vo[r]  = (1.0f - z) * vd;
        io[r]  = io[r] * 0.8f + co[r];
        acc[r] += z * w;
      }
    }
    __syncthreads();                       // sZ3 consumed; loop
  }

  if (wave == 0 && (lane & 15) < ODIM) {
    const int mrow = (lane & 16) ? 8 : 0;
#pragma unroll
    for (int r = 0; r < 8; ++r)
      out[(size_t)(b0 + mrow + r) * ODIM + (lane & 15)] = acc[r];
  }
}

// ---------------------------------------------------------------------------
// Prep: pack W (NxK, f32) into f16 B-fragment layout with BN scales folded.
//   out[((nt*KS + ks)*32 + lane)*16 + j] = W[n,k] * s_out[n] * (s_in[k]?)
// ---------------------------------------------------------------------------
__global__ void pack_w_kernel(const float* __restrict__ W, int N, int K,
                              const float* __restrict__ ig,   // input BN g (or null)
                              const float* __restrict__ iv,   // input BN v
                              const float* __restrict__ og,   // output BN g
                              const float* __restrict__ ov,   // output BN v
                              _Float16* __restrict__ outp) {
  const int idx = blockIdx.x * 256 + threadIdx.x;
  if (idx >= N * K) return;
  const int KS   = K / 32;
  const int j    = idx & 15;
  const int lane = (idx >> 4) & 31;
  const int ks   = (idx >> 9) % KS;
  const int nt   = idx / (KS * 512);
  const int n = nt * 16 + (lane & 15);
  const int k = ks * 32 + ((lane & 16) ? 8 : 0) + ((j < 8) ? j : j + 8);
  float w = W[(size_t)n * K + k] * (og[n] * rsqrtf(ov[n] + BN_EPS));
  if (ig) w *= ig[k] * rsqrtf(iv[k] + BN_EPS);
  outp[idx] = (_Float16)w;
}

// Pack Wout (2x256) into one zero-padded 16-col B tile + padded bias + decay.
__global__ void pack_wout_kernel(const float* __restrict__ Wout,
                                 const float* __restrict__ bout,
                                 _Float16* __restrict__ woutp,
                                 float* __restrict__ biasOut,
                                 float* __restrict__ decay) {
  const int idx = blockIdx.x * 256 + threadIdx.x;
  if (idx < 16) biasOut[idx] = (idx < ODIM) ? bout[idx] : 0.0f;
  if (idx < TDIM) decay[idx] = powf(0.995f, (float)(TDIM - 1 - idx));
  if (idx >= 16 * H3) return;             // 4096 packed halves
  const int j    = idx & 15;
  const int lane = (idx >> 4) & 31;
  const int ks   = (idx >> 9);
  const int n = lane & 15;
  const int k = ks * 32 + ((lane & 16) ? 8 : 0) + ((j < 8) ? j : j + 8);
  woutp[idx] = (_Float16)((n < ODIM) ? Wout[n * H3 + k] : 0.0f);
}

// One-time x transpose: xT[t][b][f] = f16(x[b][f][t]).
__global__ void transpose_x_kernel(const float* __restrict__ x,
                                   _Float16* __restrict__ xT) {
  const size_t idx = (size_t)blockIdx.x * 256 + threadIdx.x;  // over B*F
  if (idx >= (size_t)BATCH * FDIM) return;
  const float* src = x + idx * TDIM;          // 400B-aligned
  _Float16* dst = xT + idx;                   // plane stride B*F
  const size_t S = (size_t)BATCH * FDIM;
#pragma unroll 4
  for (int t = 0; t < TDIM; t += 4) {
    float4 v4 = *(const float4*)(src + t);
    dst[(size_t)(t + 0) * S] = (_Float16)v4.x;
    dst[(size_t)(t + 1) * S] = (_Float16)v4.y;
    dst[(size_t)(t + 2) * S] = (_Float16)v4.z;
    dst[(size_t)(t + 3) * S] = (_Float16)v4.w;
  }
}

// bias1[n] = (b1[n] + sum_f W1[n,f]*t_in[f]) * s1[n] + t1[n]
__global__ void bias1_kernel(const float* __restrict__ W1, const float* __restrict__ b1,
                             const float* g1, const float* bb1, const float* m1, const float* vv1,
                             const float* gi, const float* bi, const float* mi, const float* vi,
                             float* __restrict__ outb) {
  const int n = blockIdx.x * 256 + threadIdx.x;
  if (n >= H1) return;
  const float s1 = g1[n] * rsqrtf(vv1[n] + BN_EPS);
  const float t1 = bb1[n] - m1[n] * s1;
  float sum = 0.0f;
  for (int f = 0; f < FDIM; ++f) {
    const float sif = gi[f] * rsqrtf(vi[f] + BN_EPS);
    const float tif = bi[f] - mi[f] * sif;
    sum += W1[(size_t)n * FDIM + f] * tif;
  }
  outb[n] = (b1[n] + sum) * s1 + t1;
}

// bias[n] = b[n]*s[n] + (beta[n] - m[n]*s[n])
__global__ void bias_simple_kernel(const float* __restrict__ b, const float* g,
                                   const float* bb, const float* m, const float* v,
                                   float* __restrict__ outb, int N) {
  const int n = blockIdx.x * 256 + threadIdx.x;
  if (n >= N) return;
  const float s = g[n] * rsqrtf(v[n] + BN_EPS);
  outb[n] = b[n] * s + bb[n] - m[n] * s;
}

__global__ void zero_kernel(float* __restrict__ p, size_t n) {
  size_t i = (size_t)blockIdx.x * 256 + threadIdx.x;
  if (i < n) p[i] = 0.0f;
}

// ---------------------------------------------------------------------------
extern "C" void kernel_launch(void* const* d_in, const int* in_sizes, int n_in,
                              void* d_out, int out_size, void* d_ws, size_t ws_size,
                              hipStream_t stream) {
  const float* x     = (const float*)d_in[0];
  const float* bni_g = (const float*)d_in[1];
  const float* bni_b = (const float*)d_in[2];
  const float* bni_m = (const float*)d_in[3];
  const float* bni_v = (const float*)d_in[4];
  const float* W1    = (const float*)d_in[5];
  const float* b1    = (const float*)d_in[6];
  const float* bn1_g = (const float*)d_in[7];
  const float* bn1_b = (const float*)d_in[8];
  const float* bn1_m = (const float*)d_in[9];
  const float* bn1_v = (const float*)d_in[10];
  const float* W2    = (const float*)d_in[11];
  const float* b2    = (const float*)d_in[12];
  const float* bn2_g = (const float*)d_in[13];
  const float* bn2_b = (const float*)d_in[14];
  const float* bn2_m = (const float*)d_in[15];
  const float* bn2_v = (const float*)d_in[16];
  const float* W3    = (const float*)d_in[17];
  const float* b3    = (const float*)d_in[18];
  const float* bn3_g = (const float*)d_in[19];
  const float* bn3_b = (const float*)d_in[20];
  const float* bn3_m = (const float*)d_in[21];
  const float* bn3_v = (const float*)d_in[22];
  const float* Wout  = (const float*)d_in[23];
  const float* bout  = (const float*)d_in[24];

  // -------- workspace layout --------
  _Float16* W1p   = (_Float16*)d_ws;                 // 512*1024 halves
  _Float16* W2p   = W1p + (size_t)FDIM * H1;         // 1024*512 halves
  _Float16* W3p   = W2p + (size_t)H1 * H2;           // 512*256 halves
  _Float16* Woutp = W3p + (size_t)H2 * H3;           // 16*256 halves (padded)
  float* bias1   = (float*)(Woutp + 16 * H3);
  float* bias2   = bias1 + H1;
  float* bias3   = bias2 + H2;
  float* biasOut = bias3 + H3;
  float* decay   = biasOut + 16;
  float* v1s = decay + ((TDIM + 3) & ~3);
  float* i1s = v1s + (size_t)BATCH * H1;
  float* v2s = i1s + (size_t)BATCH * H1;
  float* i2s = v2s + (size_t)BATCH * H2;
  float* v3s = i2s + (size_t)BATCH * H2;
  float* i3s = v3s + (size_t)BATCH * H3;
  const size_t stateFloats = (size_t)BATCH * (2 * H1 + 2 * H2 + 2 * H3);
  _Float16* xT = (_Float16*)(i3s + (size_t)BATCH * H3);
  const size_t fixedBytes = (size_t)((char*)xT - (char*)d_ws);
  const size_t xTBytes = (size_t)TDIM * BATCH * FDIM * sizeof(_Float16);
  const int use_xT = (ws_size >= fixedBytes + xTBytes) ? 1 : 0;

  // -------- prep: fold BN, pack weights into WMMA B-fragment layout --------
  pack_w_kernel<<<(FDIM * H1 + 255) / 256, 256, 0, stream>>>(
      W1, H1, FDIM, bni_g, bni_v, bn1_g, bn1_v, W1p);
  pack_w_kernel<<<(H1 * H2 + 255) / 256, 256, 0, stream>>>(
      W2, H2, H1, nullptr, nullptr, bn2_g, bn2_v, W2p);
  pack_w_kernel<<<(H2 * H3 + 255) / 256, 256, 0, stream>>>(
      W3, H3, H2, nullptr, nullptr, bn3_g, bn3_v, W3p);
  pack_wout_kernel<<<(16 * H3 + 255) / 256, 256, 0, stream>>>(
      Wout, bout, Woutp, biasOut, decay);
  bias1_kernel<<<(H1 + 255) / 256, 256, 0, stream>>>(
      W1, b1, bn1_g, bn1_b, bn1_m, bn1_v, bni_g, bni_b, bni_m, bni_v, bias1);
  bias_simple_kernel<<<(H2 + 255) / 256, 256, 0, stream>>>(
      b2, bn2_g, bn2_b, bn2_m, bn2_v, bias2, H2);
  bias_simple_kernel<<<(H3 + 255) / 256, 256, 0, stream>>>(
      b3, bn3_g, bn3_b, bn3_m, bn3_v, bias3, H3);
  zero_kernel<<<(unsigned)((stateFloats + 255) / 256), 256, 0, stream>>>(v1s, stateFloats);
  if (use_xT) {
    transpose_x_kernel<<<(BATCH * FDIM + 255) / 256, 256, 0, stream>>>(x, xT);
  }

  // -------- persistent fused SNN kernel --------
  const int smemBytes = (M_TILE * (2 * FDIM + H1 + H2 + H3)) * (int)sizeof(_Float16); // 88 KB
  (void)hipFuncSetAttribute((const void*)snn_main,
                            hipFuncAttributeMaxDynamicSharedMemorySize, smemBytes);
  snn_main<<<BATCH / M_TILE, NTHREADS, smemBytes, stream>>>(
      x, xT, use_xT, W1p, W2p, W3p, Woutp, bias1, bias2, bias3, biasOut, decay,
      v1s, i1s, v2s, i2s, v3s, i3s, (float*)d_out);
}
